// Attention_7945689498171
// MI455X (gfx1250) — matrix operand order repaired and checked by reading, not compiled, not run on hardware
//
#include <hip/hip_runtime.h>

typedef __attribute__((ext_vector_type(16))) __bf16 v16bf;
typedef __attribute__((ext_vector_type(8)))  __bf16 v8bf;
typedef __attribute__((ext_vector_type(8)))  float  v8f;
typedef __attribute__((ext_vector_type(4)))  float  v4f;
typedef int v4i __attribute__((vector_size(16)));

#define DI __device__ __forceinline__
#define AS1 __attribute__((address_space(1)))
#define AS3 __attribute__((address_space(3)))

static constexpr int Bsz = 2, T = 2048, C = 1024, H = 16, Dh = 64;

// ---------------------------------------------------------------------------
// Fragment helpers (CDNA5 ISA 7.12.2, wave32). All loads are 2 x b128.
// A (16x32 bf16): lane half h, row r=lane&15: elem e = row[h*16+e]
//                 (contiguous run of 16)
// B (32x16 bf16): lane n=lane&15, half h: elem e = B[h*16+e][n]
// C/D (16x16 f32): col = lane&15, row = vgpr + (lane>=16 ? 8 : 0)
// ---------------------------------------------------------------------------
DI v16bf frag2(v8bf lo, v8bf hi) {
  union { v16bf v; v8bf h[2]; } u;
  u.h[0] = lo; u.h[1] = hi;
  return u.v;
}

DI v16bf load_a_frag(const __bf16* tile, int ld, int lane) {
  const __bf16* row = tile + (lane & 15) * ld + (lane >> 4) * 16;
  return frag2(*(const v8bf*)row, *(const v8bf*)(row + 8));
}

DI v16bf load_seq_frag(const __bf16* p) {   // 16 contiguous bf16, 16B aligned
  return frag2(*(const v8bf*)p, *(const v8bf*)(p + 8));
}

// async global->LDS b128 (tracked by ASYNCcnt); builtin takes typed AS pointers
DI void async_copy_b128(const __bf16* g, __bf16* l) {
#if __has_builtin(__builtin_amdgcn_global_load_async_to_lds_b128)
  __builtin_amdgcn_global_load_async_to_lds_b128((AS1 v4i*)g, (AS3 v4i*)l, 0, 0);
#else
  *(v8bf*)l = *(const v8bf*)g;
#endif
}

DI void wait_async0() {
#if __has_builtin(__builtin_amdgcn_s_wait_asynccnt)
  __builtin_amdgcn_s_wait_asynccnt(0);
#else
  asm volatile("s_wait_asynccnt 0" ::: "memory");
#endif
}

DI v8bf cvt8(v4f a, v4f b) {
  v8bf o;
#pragma unroll
  for (int i = 0; i < 4; ++i) { o[i] = (__bf16)a[i]; o[4 + i] = (__bf16)b[i]; }
  return o;
}

// ---------------------------------------------------------------------------
// Kernel 1: fused Q/K/V projection  out = x @ W.T
//   Q,K -> (B,H,T,Dh) bf16 ; V -> (B,H,Dh,T) bf16 (pre-transposed for PV)
// block = 128 threads (4 waves), 64x64 tile, K-step 32
// ---------------------------------------------------------------------------
__global__ __launch_bounds__(128)
void proj_qkv_kernel(const float* __restrict__ X,
                     const float* __restrict__ Wq,
                     const float* __restrict__ Wk,
                     const float* __restrict__ Wv,
                     __bf16* __restrict__ Qo,
                     __bf16* __restrict__ Ko,
                     __bf16* __restrict__ Vo) {
  __shared__ __align__(16) __bf16 Xs[64][40];        // A tile, natural layout
  __shared__ __align__(16) __bf16 BsF[4][32][16];    // B tile, fragment-major
  const int tid = threadIdx.x;
  const int lane = tid & 31, wave = tid >> 5;
  const int n0 = blockIdx.x * 64, m0 = blockIdx.y * 64;
  const float* W   = (blockIdx.z == 0) ? Wq : (blockIdx.z == 1) ? Wk : Wv;
  __bf16*      dst = (blockIdx.z == 0) ? Qo : (blockIdx.z == 1) ? Ko : Vo;

  v8f acc[4];
#pragma unroll
  for (int i = 0; i < 4; ++i) acc[i] = {};

  for (int k0 = 0; k0 < C; k0 += 32) {
    __syncthreads();
#pragma unroll
    for (int i = 0; i < 2; ++i) {           // X tile 64x32: 8 f32 per chunk
      int chunk = tid + i * 128;            // 256 chunks
      int r = chunk >> 2, c8 = (chunk & 3) * 8;
      const float* src = &X[(size_t)(m0 + r) * C + (k0 + c8)];
      *(v8bf*)&Xs[r][c8] = cvt8(*(const v4f*)src, *(const v4f*)(src + 4));
    }
#pragma unroll
    for (int i = 0; i < 2; ++i) {           // W^T tile 32x64 -> fragment-major
      int chunk = tid + i * 128;
      int c = chunk >> 2, r8 = (chunk & 3) * 8;       // r8 in {0,8,16,24}
      const float* src = &W[(size_t)(n0 + c) * C + (k0 + r8)];
      *(v8bf*)&BsF[c >> 4][(r8 & 16) + (c & 15)][r8 & 15] =
          cvt8(*(const v4f*)src, *(const v4f*)(src + 4));
    }
    __syncthreads();
    v16bf a = load_a_frag(&Xs[wave * 16][0], 40, lane);
#pragma unroll
    for (int nd = 0; nd < 4; ++nd) {
      v16bf b = load_seq_frag(&BsF[nd][lane][0]);
      acc[nd] = __builtin_amdgcn_wmma_f32_16x16x32_bf16(
          false, a, false, b, (short)0, acc[nd], false, false);
    }
  }
  const int half = lane >> 4, nn = lane & 15;
  const bool isV = (blockIdx.z == 2);
#pragma unroll
  for (int nd = 0; nd < 4; ++nd)
#pragma unroll
    for (int jv = 0; jv < 8; ++jv) {
      int m  = m0 + wave * 16 + jv + half * 8;
      int ng = n0 + nd * 16 + nn;
      int bb = m >> 11, t = m & (T - 1);
      int h = ng >> 6, d = ng & 63;
      size_t idx = isV ? ((((size_t)bb * H + h) * Dh + d) * T + t)   // (B,H,Dh,T)
                       : ((((size_t)bb * H + h) * T + t) * Dh + d);  // (B,H,T,Dh)
      dst[idx] = (__bf16)acc[nd][jv];
    }
}

// ---------------------------------------------------------------------------
// Kernel 2: RoPE in place on Q and K ((B,H,T,Dh) bf16), 8-wide vectors
// ---------------------------------------------------------------------------
__global__ __launch_bounds__(256)
void rope_kernel(__bf16* __restrict__ Q, __bf16* __restrict__ K,
                 const float* __restrict__ cosT, const float* __restrict__ sinT) {
  int idx = blockIdx.x * 256 + threadIdx.x;   // B*H*T*4 threads
  int d0  = (idx & 3) * 8;
  int t   = (idx >> 2) & (T - 1);
  int bh  = idx >> 13;
  size_t base = ((size_t)bh * T + t) * Dh;
  v4f c0a = *(const v4f*)&cosT[t * Dh + d0];
  v4f c0b = *(const v4f*)&cosT[t * Dh + d0 + 4];
  v4f s0a = *(const v4f*)&sinT[t * Dh + d0];
  v4f s0b = *(const v4f*)&sinT[t * Dh + d0 + 4];
  v4f c1a = *(const v4f*)&cosT[t * Dh + d0 + 32];
  v4f c1b = *(const v4f*)&cosT[t * Dh + d0 + 36];
  v4f s1a = *(const v4f*)&sinT[t * Dh + d0 + 32];
  v4f s1b = *(const v4f*)&sinT[t * Dh + d0 + 36];
#pragma unroll
  for (int w = 0; w < 2; ++w) {
    __bf16* P = w ? K : Q;
    v8bf lo = *(const v8bf*)&P[base + d0];
    v8bf hi = *(const v8bf*)&P[base + d0 + 32];
    v8bf olo, ohi;
#pragma unroll
    for (int i = 0; i < 8; ++i) {
      float c0 = (i < 4) ? c0a[i] : c0b[i - 4], s0 = (i < 4) ? s0a[i] : s0b[i - 4];
      float c1 = (i < 4) ? c1a[i] : c1b[i - 4], s1 = (i < 4) ? s1a[i] : s1b[i - 4];
      float lv = (float)lo[i], hv = (float)hi[i];
      olo[i] = (__bf16)(lv * c0 - hv * s0);
      ohi[i] = (__bf16)(hv * c1 + lv * s1);
    }
    *(v8bf*)&P[base + d0]      = olo;
    *(v8bf*)&P[base + d0 + 32] = ohi;
  }
}

// ---------------------------------------------------------------------------
// Kernel 3: causal flash attention. 1 block = (b,h, 64 q rows), 4 waves.
// K tile (natural) and V tile (transposed) async-copied to LDS.
// ---------------------------------------------------------------------------
__global__ __launch_bounds__(128)
void flash_attn_kernel(const __bf16* __restrict__ Q,
                       const __bf16* __restrict__ K,
                       const __bf16* __restrict__ Vt,   // (B,H,Dh,T)
                       __bf16* __restrict__ O) {
  __shared__ __align__(16) __bf16 Ks[32][72];     // [key s][d]
  __shared__ __align__(16) __bf16 VsT[64][40];    // [d][key s]
  __shared__ __align__(16) __bf16 Ps[4][16][40];  // per-wave P, A-layout rows
  const int tid = threadIdx.x, lane = tid & 31, wave = tid >> 5;
  const int half = lane >> 4, nn = lane & 15;
  const int qt = blockIdx.x;
  const int bh = blockIdx.y;
  const int bb = bh >> 4, h = bh & 15;
  const __bf16* Qb  = Q  + (size_t)bh * T * Dh;
  const __bf16* Kb  = K  + (size_t)bh * T * Dh;
  const __bf16* Vtb = Vt + (size_t)bh * Dh * T;
  const int q0w = qt * 64 + wave * 16;
  const float scale = 0.125f;                 // 1/sqrt(64)

  v16bf Qa0 = load_a_frag(Qb + (size_t)q0w * Dh, Dh, lane);        // d 0..31
  v16bf Qa1 = load_a_frag(Qb + (size_t)q0w * Dh + 32, Dh, lane);   // d 32..63

  v8f Oacc[4];
#pragma unroll
  for (int i = 0; i < 4; ++i) Oacc[i] = {};
  float mrow[8], lrow[8];
#pragma unroll
  for (int i = 0; i < 8; ++i) { mrow[i] = -3.0e38f; lrow[i] = 0.0f; }

  const int jEnd = qt * 64 + 64;
  for (int j = 0; j < jEnd; j += 32) {
    __syncthreads();
#pragma unroll
    for (int i = 0; i < 2; ++i) {             // K tile 32x64, b128 chunks
      int chunk = tid + i * 128;              // 256 chunks
      int r = chunk >> 3, c8 = (chunk & 7) * 8;
      async_copy_b128(Kb + (size_t)(j + r) * Dh + c8, &Ks[r][c8]);
    }
#pragma unroll
    for (int i = 0; i < 2; ++i) {             // V^T tile 64x32, b128 chunks
      int chunk = tid + i * 128;
      int d = chunk >> 2, s8 = (chunk & 3) * 8;
      async_copy_b128(Vtb + (size_t)d * T + (j + s8), &VsT[d][s8]);
    }
    if (j + 32 < jEnd) {                      // prefetch next K/V tiles
      __builtin_prefetch(Kb + (size_t)(j + 32) * Dh + tid * 16, 0, 1);
      __builtin_prefetch(Vtb + (size_t)(tid >> 1) * T + j + 32, 0, 1);
    }
    wait_async0();
    __syncthreads();
    if (j > q0w + 15) continue;               // tile fully above causal diag

    // S = Q * K^T : B frag is contiguous along d in Ks rows
    v8f S0 = {}, S1 = {};
    S0 = __builtin_amdgcn_wmma_f32_16x16x32_bf16(
        false, Qa0, false, load_seq_frag(&Ks[nn][half * 16]), (short)0, S0, false, false);
    S0 = __builtin_amdgcn_wmma_f32_16x16x32_bf16(
        false, Qa1, false, load_seq_frag(&Ks[nn][32 + half * 16]), (short)0, S0, false, false);
    S1 = __builtin_amdgcn_wmma_f32_16x16x32_bf16(
        false, Qa0, false, load_seq_frag(&Ks[16 + nn][half * 16]), (short)0, S1, false, false);
    S1 = __builtin_amdgcn_wmma_f32_16x16x32_bf16(
        false, Qa1, false, load_seq_frag(&Ks[16 + nn][32 + half * 16]), (short)0, S1, false, false);

    // online softmax: rows live in 16-lane groups, reduce with shfl_xor <= 8
#pragma unroll
    for (int jv = 0; jv < 8; ++jv) {
      int Mr = jv + half * 8;
      int qi = q0w + Mr;
      float s0 = S0[jv] * scale;
      float s1 = S1[jv] * scale;
      if (j + nn > qi)      s0 = -3.0e38f;
      if (j + 16 + nn > qi) s1 = -3.0e38f;
      float rm = fmaxf(s0, s1);
#pragma unroll
      for (int off = 1; off < 16; off <<= 1)
        rm = fmaxf(rm, __shfl_xor(rm, off, 32));
      float mnew  = fmaxf(mrow[jv], rm);
      float alpha = __expf(mrow[jv] - mnew);
      float p0 = __expf(s0 - mnew);
      float p1 = __expf(s1 - mnew);
      float rs = p0 + p1;
#pragma unroll
      for (int off = 1; off < 16; off <<= 1)
        rs += __shfl_xor(rs, off, 32);
      lrow[jv] = lrow[jv] * alpha + rs;
      mrow[jv] = mnew;
#pragma unroll
      for (int nd = 0; nd < 4; ++nd) Oacc[nd][jv] *= alpha;
      Ps[wave][Mr][nn]      = (__bf16)p0;     // C-layout -> LDS -> A-layout
      Ps[wave][Mr][16 + nn] = (__bf16)p1;
    }

    // O += P * V : B frag contiguous along s in VsT rows
    v16bf Pa = load_a_frag(&Ps[wave][0][0], 40, lane);
#pragma unroll
    for (int nd = 0; nd < 4; ++nd) {
      v16bf bv = load_seq_frag(&VsT[nd * 16 + nn][half * 16]);
      Oacc[nd] = __builtin_amdgcn_wmma_f32_16x16x32_bf16(
          false, Pa, false, bv, (short)0, Oacc[nd], false, false);
    }
  }

  // normalize and store to (B,T,H,Dh)
#pragma unroll
  for (int jv = 0; jv < 8; ++jv) {
    int Mr = jv + half * 8;
    int t  = q0w + Mr;
    float inv = 1.0f / lrow[jv];
#pragma unroll
    for (int nd = 0; nd < 4; ++nd) {
      int d = nd * 16 + nn;
      O[(((size_t)bb * T + t) * H + h) * Dh + d] = (__bf16)(Oacc[nd][jv] * inv);
    }
  }
}

// ---------------------------------------------------------------------------
// Kernel 4: output projection  out = attOut @ Wo.T  (f32 epilogue)
// activation tile async-copied (already bf16); weights converted in regs
// ---------------------------------------------------------------------------
__global__ __launch_bounds__(128)
void proj_out_kernel(const __bf16* __restrict__ A,   // (B*T, C) bf16
                     const float* __restrict__ Wo,
                     float* __restrict__ out) {
  __shared__ __align__(16) __bf16 Xs[64][40];
  __shared__ __align__(16) __bf16 BsF[4][32][16];
  const int tid = threadIdx.x;
  const int lane = tid & 31, wave = tid >> 5;
  const int n0 = blockIdx.x * 64, m0 = blockIdx.y * 64;

  v8f acc[4];
#pragma unroll
  for (int i = 0; i < 4; ++i) acc[i] = {};

  for (int k0 = 0; k0 < C; k0 += 32) {
    __syncthreads();
#pragma unroll
    for (int i = 0; i < 2; ++i) {             // activation tile: async b128
      int chunk = tid + i * 128;
      int r = chunk >> 2, c8 = (chunk & 3) * 8;
      async_copy_b128(&A[(size_t)(m0 + r) * C + (k0 + c8)], &Xs[r][c8]);
    }
#pragma unroll
    for (int i = 0; i < 2; ++i) {             // Wo^T -> fragment-major bf16
      int chunk = tid + i * 128;
      int c = chunk >> 2, r8 = (chunk & 3) * 8;
      const float* src = &Wo[(size_t)(n0 + c) * C + (k0 + r8)];
      *(v8bf*)&BsF[c >> 4][(r8 & 16) + (c & 15)][r8 & 15] =
          cvt8(*(const v4f*)src, *(const v4f*)(src + 4));
    }
    wait_async0();
    __syncthreads();
    v16bf a = load_a_frag(&Xs[wave * 16][0], 40, lane);
#pragma unroll
    for (int nd = 0; nd < 4; ++nd) {
      v16bf b = load_seq_frag(&BsF[nd][lane][0]);
      acc[nd] = __builtin_amdgcn_wmma_f32_16x16x32_bf16(
          false, a, false, b, (short)0, acc[nd], false, false);
    }
  }
  const int half = lane >> 4, nn = lane & 15;
#pragma unroll
  for (int nd = 0; nd < 4; ++nd)
#pragma unroll
    for (int jv = 0; jv < 8; ++jv) {
      int m  = m0 + wave * 16 + jv + half * 8;
      int ng = n0 + nd * 16 + nn;
      out[(size_t)m * C + ng] = acc[nd][jv];
    }
}

// ---------------------------------------------------------------------------
// Launch
// ---------------------------------------------------------------------------
extern "C" void kernel_launch(void* const* d_in, const int* in_sizes, int n_in,
                              void* d_out, int out_size, void* d_ws, size_t ws_size,
                              hipStream_t stream) {
  const float* x    = (const float*)d_in[0];
  const float* cosT = (const float*)d_in[1];
  const float* sinT = (const float*)d_in[2];
  // d_in[3] = mask (causality applied analytically), d_in[4] = n_heads (=16)
  const float* Wq = (const float*)d_in[5];
  const float* Wk = (const float*)d_in[6];
  const float* Wv = (const float*)d_in[7];
  const float* Wo = (const float*)d_in[8];
  float* out = (float*)d_out;

  // workspace: Q | K | Vt | attOut, each B*T*C bf16 = 8 MiB
  const size_t seg = (size_t)Bsz * H * T * Dh * sizeof(__bf16);
  char* ws = (char*)d_ws;
  __bf16* Qw  = (__bf16*)(ws + 0 * seg);
  __bf16* Kw  = (__bf16*)(ws + 1 * seg);
  __bf16* Vtw = (__bf16*)(ws + 2 * seg);
  __bf16* Aw  = (__bf16*)(ws + 3 * seg);

  proj_qkv_kernel<<<dim3(C / 64, (Bsz * T) / 64, 3), 128, 0, stream>>>(
      x, Wq, Wk, Wv, Qw, Kw, Vtw);

  rope_kernel<<<(Bsz * H * T * 4) / 256, 256, 0, stream>>>(Qw, Kw, cosT, sinT);

  flash_attn_kernel<<<dim3(T / 64, Bsz * H), 128, 0, stream>>>(Qw, Kw, Vtw, Aw);

  proj_out_kernel<<<dim3(C / 64, (Bsz * T) / 64), 128, 0, stream>>>(Aw, Wo, out);
}
